// CEINN_67138928771458
// MI455X (gfx1250) — matrix operand order repaired and checked by reading, not compiled, not run on hardware
//
#include <hip/hip_runtime.h>
#include <hip/hip_bf16.h>
#include <math.h>

typedef _Float16 half_t;
typedef __attribute__((ext_vector_type(16))) _Float16 v16h;
typedef __attribute__((ext_vector_type(2)))  _Float16 h2;
typedef __attribute__((ext_vector_type(8)))  float    v8f;
typedef __attribute__((ext_vector_type(4)))  unsigned int u32x4;

#define B_   64
#define S_   512
#define C_   200
#define D_   256
#define H_   4
#define DH_  64
#define SEM_ 128
#define ECO_ 128
#define P_   8

static __device__ __forceinline__ float gelu_exact(float x) {
  return 0.5f * x * (1.0f + erff(x * 0.70710678118654752f));
}

// Fragment loads as 2x ds_load_b128 (rows padded so chunks are 16B aligned).
// A-frag (ISA 7.12.2, 16-bit A 16x32): lane halves [g*8, g*8+7] and [g*8+16, g*8+23].
// B-frag (16-bit B 32x16): lane halves [g*8, g*8+7] and [g*8+16, g*8+23].
static __device__ __forceinline__ v16h load_frag_a(const half_t* rowp, int g) {
  union { v16h v; u32x4 q[2]; } u;
  u.q[0] = *(const u32x4*)(rowp + g * 8);
  u.q[1] = *(const u32x4*)(rowp + g * 8 + 16);
  return u.v;
}
static __device__ __forceinline__ v16h load_frag_b(const half_t* rowp, int g) {
  union { v16h v; u32x4 q[2]; } u;
  u.q[0] = *(const u32x4*)(rowp + g * 8);
  u.q[1] = *(const u32x4*)(rowp + g * 8 + 16);
  return u.v;
}

// --------------------------------------------------------------------------
// Generic WMMA GEMM: Y[M,N] = act(X[M,K] @ W[K,N] + bias + res)
// Block: 256 threads (8 waves), tile 128(M) x 64(N), K step 64 (K % 64 == 0).
// 8x v_wmma_f32_16x16x32_f16 per wave per K step, grouped loads -> one wait
// per 4 WMMAs.
// --------------------------------------------------------------------------
template<int ACT>
__global__ __launch_bounds__(256) void gemm_wmma(
    const float* __restrict__ X, const float* __restrict__ W,
    const float* __restrict__ bias, const float* __restrict__ res,
    float* __restrict__ Y, int M, int N, int K)
{
  __shared__ alignas(16) half_t As[128][72];   // stride 144B: 16B-aligned chunks
  __shared__ alignas(16) half_t Bs[64][72];
  const int tid  = threadIdx.x;
  const int lane = tid & 31;
  const int wave = tid >> 5;
  const int g = lane >> 4;
  const int r = lane & 15;
  const int m0 = blockIdx.x * 128;
  const int n0 = blockIdx.y * 64;

  v8f acc[4] = {};

  for (int k0 = 0; k0 < K; k0 += 64) {
    __syncthreads();
    // stage A: 2 consecutive k per thread -> packed cvt + ds_store_b32
    for (int i = tid; i < 128 * 32; i += 256) {
      int row = i >> 5;
      int kk  = (i & 31) * 2;
      float x0 = 0.f, x1 = 0.f;
      if (m0 + row < M) {
        const float* xp = &X[(size_t)(m0 + row) * K + (k0 + kk)];
        x0 = xp[0]; x1 = xp[1];
        if (k0 + 64 < K) __builtin_prefetch(xp + 64, 0, 1);
      }
      h2 hv; hv[0] = (half_t)x0; hv[1] = (half_t)x1;
      *(h2*)&As[row][kk] = hv;
    }
    // stage B transposed (Bs[n][k]): 2 consecutive k adjacent in LDS -> packed
    for (int i = tid; i < 64 * 32; i += 256) {
      int nn = i & 63;
      int kk = (i >> 6) * 2;
      float w0 = W[(size_t)(k0 + kk)     * N + (n0 + nn)];
      float w1 = W[(size_t)(k0 + kk + 1) * N + (n0 + nn)];
      h2 hv; hv[0] = (half_t)w0; hv[1] = (half_t)w1;
      *(h2*)&Bs[nn][kk] = hv;
    }
    __syncthreads();

    #pragma unroll
    for (int kh = 0; kh < 2; ++kh) {
      v16h a = load_frag_a(&As[wave * 16 + r][kh * 32], g);
      v16h bf[4];
      #pragma unroll
      for (int j = 0; j < 4; ++j)
        bf[j] = load_frag_b(&Bs[j * 16 + r][kh * 32], g);
      #pragma unroll
      for (int j = 0; j < 4; ++j)
        acc[j] = __builtin_amdgcn_wmma_f32_16x16x32_f16(
            false, a, false, bf[j], (short)0, acc[j], false, false);
    }
  }

  #pragma unroll
  for (int j = 0; j < 4; ++j) {
    #pragma unroll
    for (int e = 0; e < 8; ++e) {
      int row = m0 + wave * 16 + e + 8 * g;
      int col = n0 + j * 16 + r;
      if (row < M) {
        float val = acc[j][e];
        if (bias) val += bias[col];
        if (res)  val += res[(size_t)row * N + col];
        if (ACT == 1) val = gelu_exact(val);
        Y[(size_t)row * N + col] = val;
      }
    }
  }
}

// --------------------------------------------------------------------------
// Fused flash attention with hyperbolic-discount bias.
// Grid: (S/64, H, B), block 128 (4 waves). Wave w handles 16 query rows.
// K/V chunks of 32 keys; online softmax; never materializes SxS.
// --------------------------------------------------------------------------
__global__ __launch_bounds__(128) void flash_attn_kernel(
    const float* __restrict__ Q, const float* __restrict__ Kin,
    const float* __restrict__ Vin, const float* __restrict__ gamma,
    float* __restrict__ O)
{
  __shared__ alignas(16) half_t Ql[64][40];       // [qrow][feat]
  __shared__ alignas(16) half_t Kl[32][72];       // [key][feat 0..63]
  __shared__ alignas(16) half_t Vl[64][40];       // [feat][key]
  __shared__ alignas(16) half_t Pl[4][16][40];    // per-wave P tile [qrow][key]

  const int lane = threadIdx.x & 31;
  const int wave = threadIdx.x >> 5;
  const int g = lane >> 4, r = lane & 15;
  const int qb0 = blockIdx.x * 64;
  const int h   = blockIdx.y;
  const int b   = blockIdx.z;
  const size_t base = (size_t)b * S_ * D_ + (size_t)h * DH_;

  // stage Q (pre-scaled by 1/sqrt(DH)) via LDS, then pin fragments in regs
  for (int i = threadIdx.x; i < 64 * 32; i += 128) {
    int row = i >> 5, f = (i & 31) * 2;
    const float* qp = Q + base + (size_t)(qb0 + row) * D_ + f;
    h2 hv; hv[0] = (half_t)(0.125f * qp[0]); hv[1] = (half_t)(0.125f * qp[1]);
    *(h2*)&Ql[row][f] = hv;
  }
  __syncthreads();
  const v16h aq0 = load_frag_a(&Ql[wave * 16 + r][0], g);
  const v16h aq1 = load_frag_a(&Ql[wave * 16 + r][32], g);

  float gam[8];
  #pragma unroll
  for (int e = 0; e < 8; ++e)
    gam[e] = gamma[b * S_ + qb0 + wave * 16 + e + 8 * g];

  float mi[8], li[8];
  #pragma unroll
  for (int e = 0; e < 8; ++e) { mi[e] = -3.0e38f; li[e] = 0.0f; }
  v8f o[4] = {};

  const int qmax = qb0 + 63;
  for (int j0 = 0; j0 <= qmax; j0 += 32) {
    __syncthreads();
    // K tile: packed along feat
    for (int i = threadIdx.x; i < 32 * 32; i += 128) {
      int key = i >> 5, f = (i & 31) * 2;
      const float* kp = &Kin[base + (size_t)(j0 + key) * D_ + f];
      h2 hk; hk[0] = (half_t)kp[0]; hk[1] = (half_t)kp[1];
      *(h2*)&Kl[key][f] = hk;
    }
    // V tile transposed (Vl[f][key]): 2 consecutive keys adjacent -> packed
    for (int i = threadIdx.x; i < 16 * 64; i += 128) {
      int k2 = (i >> 6) * 2, f = i & 63;
      float v0 = Vin[base + (size_t)(j0 + k2)     * D_ + f];
      float v1 = Vin[base + (size_t)(j0 + k2 + 1) * D_ + f];
      h2 hv; hv[0] = (half_t)v0; hv[1] = (half_t)v1;
      *(h2*)&Vl[f][k2] = hv;
    }
    __syncthreads();

    // scores: 16q x 32keys as two 16x16 fragments; grouped loads then 4 WMMAs
    v16h bk[4];
    #pragma unroll
    for (int st = 0; st < 2; ++st) {
      const half_t* kp = &Kl[st * 16 + r][0];
      bk[2 * st]     = load_frag_b(kp, g);
      bk[2 * st + 1] = load_frag_b(kp + 32, g);
    }
    v8f sfr[2] = {{}, {}};
    #pragma unroll
    for (int st = 0; st < 2; ++st) {
      sfr[st] = __builtin_amdgcn_wmma_f32_16x16x32_f16(
          false, aq0, false, bk[2 * st], (short)0, sfr[st], false, false);
      sfr[st] = __builtin_amdgcn_wmma_f32_16x16x32_f16(
          false, aq1, false, bk[2 * st + 1], (short)0, sfr[st], false, false);
    }
    // causal mask + hyperbolic bias -log1p(gamma*delta + 1e-8)
    #pragma unroll
    for (int st = 0; st < 2; ++st) {
      #pragma unroll
      for (int e = 0; e < 8; ++e) {
        int qrow = qb0 + wave * 16 + e + 8 * g;
        int key  = j0 + st * 16 + r;
        float sv = sfr[st][e];
        if (key > qrow) sv = -3.0e38f;
        else sv -= log1pf(gam[e] * (float)(qrow - key) + 1e-8f);
        sfr[st][e] = sv;
      }
    }
    // online softmax
    float pn0[8], pn1[8];
    #pragma unroll
    for (int e = 0; e < 8; ++e) {
      float mx = fmaxf(sfr[0][e], sfr[1][e]);
      #pragma unroll
      for (int msk = 1; msk < 16; msk <<= 1)
        mx = fmaxf(mx, __shfl_xor(mx, msk, 32));
      float mnew = fmaxf(mi[e], mx);
      float sc = __expf(mi[e] - mnew);
      float p0 = __expf(sfr[0][e] - mnew);
      float p1 = __expf(sfr[1][e] - mnew);
      float rs = p0 + p1;
      #pragma unroll
      for (int msk = 1; msk < 16; msk <<= 1)
        rs += __shfl_xor(rs, msk, 32);
      li[e] = li[e] * sc + rs;
      mi[e] = mnew;
      #pragma unroll
      for (int j = 0; j < 4; ++j) o[j][e] *= sc;
      pn0[e] = p0; pn1[e] = p1;
    }
    // redistribute P (C-layout) into A-fragment layout via LDS
    #pragma unroll
    for (int e = 0; e < 8; ++e) {
      Pl[wave][e + 8 * g][r]      = (half_t)pn0[e];
      Pl[wave][e + 8 * g][16 + r] = (half_t)pn1[e];
    }
    __syncthreads();
    v16h apf = load_frag_a(&Pl[wave][r][0], g);
    v16h bv[4];
    #pragma unroll
    for (int j = 0; j < 4; ++j)
      bv[j] = load_frag_b(&Vl[j * 16 + r][0], g);
    #pragma unroll
    for (int j = 0; j < 4; ++j)
      o[j] = __builtin_amdgcn_wmma_f32_16x16x32_f16(
          false, apf, false, bv[j], (short)0, o[j], false, false);
  }

  #pragma unroll
  for (int j = 0; j < 4; ++j) {
    #pragma unroll
    for (int e = 0; e < 8; ++e) {
      int qrow = qb0 + wave * 16 + e + 8 * g;
      O[base + (size_t)qrow * D_ + j * 16 + r] = o[j][e] / li[e];
    }
  }
}

// --------------------------------------------------------------------------
// Token embedding: out = item_table[id] (0 if id==0) + pos + side_feat[id]@side_W + side_b
// --------------------------------------------------------------------------
__global__ __launch_bounds__(256) void embed_kernel(
    const int* __restrict__ ids, const float* __restrict__ item_table,
    const float* __restrict__ pos_table, const float* __restrict__ side_feat,
    const float* __restrict__ side_W, const float* __restrict__ side_b,
    float* __restrict__ out, int Scols, int use_pos)
{
  __shared__ float sf[64];
  const int tok = blockIdx.x;
  const int t = threadIdx.x;
  const int idx = ids[tok];
  if (t < 64) sf[t] = side_feat[(size_t)idx * 64 + t];
  __syncthreads();
  float acc = side_b[t];
  if (idx != 0) acc += item_table[(size_t)idx * D_ + t];
  if (use_pos) acc += pos_table[(size_t)(tok % Scols) * D_ + t];
  #pragma unroll 8
  for (int j = 0; j < 64; ++j) acc += sf[j] * side_W[j * D_ + t];
  out[(size_t)tok * D_ + t] = acc;
}

// gamma[row] = sigmoid(eco_x[row,:128] . gamW + gamb); one wave per row
__global__ __launch_bounds__(256) void gamma_kernel(
    const float* __restrict__ eco_x, const float* __restrict__ gamW,
    const float* __restrict__ gamb, float* __restrict__ gam)
{
  const int row  = blockIdx.x * 8 + (threadIdx.x >> 5);
  const int lane = threadIdx.x & 31;
  float s = 0.f;
  for (int k = lane; k < ECO_; k += 32) s += eco_x[(size_t)row * ECO_ + k] * gamW[k];
  #pragma unroll
  for (int msk = 16; msk > 0; msk >>= 1) s += __shfl_xor(s, msk, 32);
  if (lane == 0) gam[row] = 1.f / (1.f + expf(-(s + gamb[0])));
}

// LayerNorm over D=256 (optionally with residual), in-place safe
__global__ __launch_bounds__(256) void ln_kernel(
    const float* __restrict__ a, const float* __restrict__ res,
    float* __restrict__ out, const float* __restrict__ gm,
    const float* __restrict__ bt)
{
  __shared__ float red[256];
  const int t = threadIdx.x;
  const size_t row = blockIdx.x;
  float v = a[row * D_ + t];
  if (res) v += res[row * D_ + t];
  red[t] = v; __syncthreads();
  for (int s = 128; s > 0; s >>= 1) { if (t < s) red[t] += red[t + s]; __syncthreads(); }
  float mean = red[0] * (1.0f / D_); __syncthreads();
  float d = v - mean;
  red[t] = d * d; __syncthreads();
  for (int s = 128; s > 0; s >>= 1) { if (t < s) red[t] += red[t + s]; __syncthreads(); }
  float var = red[0] * (1.0f / D_); __syncthreads();
  out[row * D_ + t] = d * rsqrtf(var + 1e-5f) * gm[t] + bt[t];
}

// last[b,:] = X[b, max(count(seq!=0),1)-1, :]
__global__ __launch_bounds__(256) void last_kernel(
    const int* __restrict__ seq, const float* __restrict__ X,
    float* __restrict__ last)
{
  __shared__ int red[256];
  const int b = blockIdx.x, t = threadIdx.x;
  int c = 0;
  for (int s = t; s < S_; s += 256) c += (seq[b * S_ + s] != 0) ? 1 : 0;
  red[t] = c; __syncthreads();
  for (int s = 128; s > 0; s >>= 1) { if (t < s) red[t] += red[t + s]; __syncthreads(); }
  int len = red[0] > 0 ? red[0] : 1;
  last[(size_t)b * D_ + t] = X[((size_t)b * S_ + len - 1) * D_ + t];
}

// do_util[b,c] = sum_p prior_p * (tanh(pre_p) . doOutW) + doOutb
__global__ __launch_bounds__(256) void do_util_kernel(
    const float* __restrict__ itemWi, const float* __restrict__ ecoWe,
    const float* __restrict__ protoWp, const float* __restrict__ doOutW,
    const float* __restrict__ doOutb, const float* __restrict__ plog,
    float* __restrict__ out_do)
{
  __shared__ float red[256];
  const int bc = blockIdx.x, b = bc / C_, t = threadIdx.x;
  float pl[P_], pmax = -3e38f, psum = 0.f;
  #pragma unroll
  for (int p = 0; p < P_; ++p) { pl[p] = plog[p]; pmax = fmaxf(pmax, pl[p]); }
  #pragma unroll
  for (int p = 0; p < P_; ++p) { pl[p] = expf(pl[p] - pmax); psum += pl[p]; }
  const float base = itemWi[(size_t)bc * D_ + t] + ecoWe[(size_t)b * D_ + t];
  float acc = 0.f;
  #pragma unroll
  for (int p = 0; p < P_; ++p)
    acc += (pl[p] / psum) * tanhf(base + protoWp[p * D_ + t]);
  red[t] = acc * doOutW[t]; __syncthreads();
  for (int s = 128; s > 0; s >>= 1) { if (t < s) red[t] += red[t + s]; __syncthreads(); }
  if (t == 0) out_do[bc] = red[0] + doOutb[0];
}

// short_s[b,c] = gelu(semWs[b]+itemWi2[bc]+shb1) . shW2 + shb2
__global__ __launch_bounds__(256) void short_kernel(
    const float* __restrict__ semWs, const float* __restrict__ itemWi2,
    const float* __restrict__ shb1, const float* __restrict__ shW2,
    const float* __restrict__ shb2, float* __restrict__ out_short)
{
  __shared__ float red[256];
  const int bc = blockIdx.x, b = bc / C_, t = threadIdx.x;
  float hv = gelu_exact(semWs[(size_t)b * D_ + t] + itemWi2[(size_t)bc * D_ + t] + shb1[t]);
  red[t] = hv * shW2[t]; __syncthreads();
  for (int s = 128; s > 0; s >>= 1) { if (t < s) red[t] += red[t + s]; __syncthreads(); }
  if (t == 0) out_short[bc] = red[0] + shb2[0];
}

// Per-batch: mean_abs, base_ref, eta, ref, ortho partial, importance weights w
__global__ __launch_bounds__(256) void perb_kernel(
    const float* __restrict__ do_out, const float* __restrict__ ecos,
    const float* __restrict__ semv, const float* __restrict__ refW,
    const float* __restrict__ refb, const float* __restrict__ etaW,
    const float* __restrict__ etab, const float* __restrict__ prop,
    const int* __restrict__ cand, float* __restrict__ out_ref,
    float* __restrict__ out_w, float* __restrict__ dsq)
{
  __shared__ float red[256];
  const int b = blockIdx.x, t = threadIdx.x;
  float v = (t < C_) ? do_out[b * C_ + t] : 0.f;
  red[t] = v; __syncthreads();
  for (int s = 128; s > 0; s >>= 1) { if (t < s) red[t] += red[t + s]; __syncthreads(); }
  float mean_abs = red[0] / (float)C_; __syncthreads();

  red[t] = (t < ECO_) ? ecos[b * ECO_ + t] * refW[t] : 0.f; __syncthreads();
  for (int s = 128; s > 0; s >>= 1) { if (t < s) red[t] += red[t + s]; __syncthreads(); }
  float base_ref = red[0] + refb[0]; __syncthreads();

  red[t] = (t < ECO_) ? ecos[b * ECO_ + t] * etaW[t] : 0.f; __syncthreads();
  for (int s = 128; s > 0; s >>= 1) { if (t < s) red[t] += red[t + s]; __syncthreads(); }
  float eta = 1.f / (1.f + expf(-(red[0] + mean_abs * etaW[ECO_] + etab[0])));
  __syncthreads();
  if (t == 0) out_ref[b] = eta * mean_abs + (1.f - eta) * base_ref;

  red[t] = (t < SEM_) ? semv[b * SEM_ + t] * ecos[b * ECO_ + t] : 0.f; __syncthreads();
  for (int s = 128; s > 0; s >>= 1) { if (t < s) red[t] += red[t + s]; __syncthreads(); }
  if (t == 0) dsq[b] = red[0] * red[0];
  __syncthreads();

  float wr = 0.f;
  if (t < C_) {
    float pr = prop[cand[b * C_ + t]];
    wr = fminf(1.f / fmaxf(pr, 1e-6f), 10.0f);
  }
  red[t] = wr; __syncthreads();
  for (int s = 128; s > 0; s >>= 1) { if (t < s) red[t] += red[t + s]; __syncthreads(); }
  float wm = red[0] / (float)C_ + 1e-8f;
  if (t < C_) out_w[b * C_ + t] = wr / wm;
}

// Prospect value + total
__global__ __launch_bounds__(256) void total_kernel(
    const float* __restrict__ do_out, const float* __restrict__ short_out,
    const float* __restrict__ ref, const float* __restrict__ kappa,
    float* __restrict__ out_total, float* __restrict__ out_long)
{
  const int i = blockIdx.x * 256 + threadIdx.x;
  if (i >= B_ * C_) return;
  const int b = i / C_;
  float du = do_out[i] - ref[b];
  float gate = 1.f / (1.f + expf(-du * 2.0f));            // GTEMP=0.5
  float q = du * du + 1e-6f;
  float gain = powf(q, 0.35f);                            // ALPHA/2
  float loss = powf(q, 0.40f);                            // BETA/2
  float ls = gate * gain - 2.0f * (1.f - gate) * loss;    // LAM=2
  float denom = 1.f + log1pf(expf(kappa[0]));             // 1+softplus
  out_long[i]  = ls;
  out_total[i] = short_out[i] + ls / denom;
}

__global__ void ortho_kernel(const float* __restrict__ dsq, float* __restrict__ out)
{
  __shared__ float red[64];
  const int t = threadIdx.x;
  red[t] = dsq[t]; __syncthreads();
  for (int s = 32; s > 0; s >>= 1) { if (t < s) red[t] += red[t + s]; __syncthreads(); }
  if (t == 0) out[0] = red[0] / (float)B_;
}

// --------------------------------------------------------------------------
extern "C" void kernel_launch(void* const* d_in, const int* in_sizes, int n_in,
                              void* d_out, int out_size, void* d_ws, size_t ws_size,
                              hipStream_t stream)
{
  (void)in_sizes; (void)n_in; (void)out_size; (void)ws_size;
  const int*   seq        = (const int*)  d_in[0];
  const int*   cand       = (const int*)  d_in[1];
  const float* item_table = (const float*)d_in[2];
  const float* pos_table  = (const float*)d_in[3];
  const float* side_feat  = (const float*)d_in[4];
  const float* prop       = (const float*)d_in[5];
  const float* side_W     = (const float*)d_in[6];
  const float* side_b     = (const float*)d_in[7];
  const float* qW  = (const float*)d_in[8];
  const float* kW  = (const float*)d_in[9];
  const float* vW  = (const float*)d_in[10];
  const float* oW  = (const float*)d_in[11];
  const float* ffW1= (const float*)d_in[12];
  const float* ffW2= (const float*)d_in[13];
  const float* qb  = (const float*)d_in[14];
  const float* kb  = (const float*)d_in[15];
  const float* vb  = (const float*)d_in[16];
  const float* ob  = (const float*)d_in[17];
  const float* ffb1= (const float*)d_in[18];
  const float* ffb2= (const float*)d_in[19];
  const float* n1g = (const float*)d_in[20];
  const float* n1b = (const float*)d_in[21];
  const float* n2g = (const float*)d_in[22];
  const float* n2b = (const float*)d_in[23];
  const float* fng = (const float*)d_in[24];
  const float* fnb = (const float*)d_in[25];
  const float* semW1=(const float*)d_in[26];
  const float* semb1=(const float*)d_in[27];
  const float* semW2=(const float*)d_in[28];
  const float* semb2=(const float*)d_in[29];
  const float* ecoW1=(const float*)d_in[30];
  const float* ecob1=(const float*)d_in[31];
  const float* ecoW2=(const float*)d_in[32];
  const float* ecob2=(const float*)d_in[33];
  const float* gamW =(const float*)d_in[34];
  const float* gamb =(const float*)d_in[35];
  const float* refW =(const float*)d_in[36];
  const float* refb =(const float*)d_in[37];
  const float* etaW =(const float*)d_in[38];
  const float* etab =(const float*)d_in[39];
  const float* shW1 =(const float*)d_in[40];
  const float* shb1 =(const float*)d_in[41];
  const float* shW2 =(const float*)d_in[42];
  const float* shb2 =(const float*)d_in[43];
  const float* doW  =(const float*)d_in[44];
  const float* dob  =(const float*)d_in[45];
  const float* doOutW=(const float*)d_in[46];
  const float* doOutb=(const float*)d_in[47];
  const float* proto=(const float*)d_in[48];
  const float* plog =(const float*)d_in[49];
  const float* kappa=(const float*)d_in[50];

  float* out = (float*)d_out;
  float* OUT_TOTAL = out;
  float* OUT_SHORT = out + 12800;
  float* OUT_LONG  = out + 25600;
  float* OUT_DO    = out + 38400;
  float* OUT_REF   = out + 51200;
  float* OUT_W     = out + 51264;
  float* OUT_ORTHO = out + 64064;

  float* ws = (float*)d_ws;
  const int    TOK = B_ * S_;                 // 32768
  const size_t XSZ = (size_t)TOK * D_;        // 8388608 floats
  float* X     = ws;  ws += XSZ;
  float* BUF_A = ws;  ws += XSZ;
  float* BUF_B = ws;  ws += XSZ;
  float* BUF_C = ws;  ws += XSZ;
  float* BUF_D = ws;  ws += XSZ;
  float* BUF_H = ws;  ws += XSZ;              // eco hidden / ffn hidden chunk
  float* ECO_X = ws;  ws += (size_t)TOK * ECO_;
  float* GAMMA = ws;  ws += TOK;
  float* LAST  = ws;  ws += B_ * D_;
  float* SB1   = ws;  ws += B_ * D_;
  float* SEMV  = ws;  ws += B_ * SEM_;
  float* ECOS  = ws;  ws += B_ * ECO_;
  float* ECOWE = ws;  ws += B_ * D_;
  float* SEMWS = ws;  ws += B_ * D_;
  float* PROTOWP = ws; ws += P_ * D_;
  float* DSQ   = ws;  ws += B_;

  auto ggrid = [](int M, int N) { return dim3((unsigned)((M + 127) / 128), (unsigned)(N / 64)); };

  // 1. token embeddings
  embed_kernel<<<TOK, 256, 0, stream>>>(seq, item_table, pos_table, side_feat,
                                        side_W, side_b, X, S_, 1);

  // 2. eco(x) -> gamma (hyperbolic discount rates)
  gemm_wmma<1><<<ggrid(TOK, D_), 256, 0, stream>>>(X, ecoW1, ecob1, nullptr, BUF_H, TOK, D_, D_);
  gemm_wmma<0><<<ggrid(TOK, ECO_), 256, 0, stream>>>(BUF_H, ecoW2, ecob2, nullptr, ECO_X, TOK, ECO_, D_);
  gamma_kernel<<<TOK / 8, 256, 0, stream>>>(ECO_X, gamW, gamb, GAMMA);

  // 3. transformer layers (fused flash attention, chunked FFN)
  for (int l = 0; l < 2; ++l) {
    gemm_wmma<0><<<ggrid(TOK, D_), 256, 0, stream>>>(X, qW + (size_t)l*D_*D_, qb + l*D_, nullptr, BUF_A, TOK, D_, D_);
    gemm_wmma<0><<<ggrid(TOK, D_), 256, 0, stream>>>(X, kW + (size_t)l*D_*D_, kb + l*D_, nullptr, BUF_B, TOK, D_, D_);
    gemm_wmma<0><<<ggrid(TOK, D_), 256, 0, stream>>>(X, vW + (size_t)l*D_*D_, vb + l*D_, nullptr, BUF_C, TOK, D_, D_);
    flash_attn_kernel<<<dim3(S_/64, H_, B_), 128, 0, stream>>>(BUF_A, BUF_B, BUF_C, GAMMA, BUF_H);
    gemm_wmma<0><<<ggrid(TOK, D_), 256, 0, stream>>>(BUF_H, oW + (size_t)l*D_*D_, ob + l*D_, nullptr, BUF_D, TOK, D_, D_);
    ln_kernel<<<TOK, 256, 0, stream>>>(X, BUF_D, X, n1g + l*D_, n1b + l*D_);
    for (int ch = 0; ch < 4; ++ch) {
      const float* xc = X + (size_t)ch * 8192 * D_;
      gemm_wmma<1><<<ggrid(8192, 4*D_), 256, 0, stream>>>(xc, ffW1 + (size_t)l*D_*4*D_, ffb1 + l*4*D_, nullptr, BUF_H, 8192, 4*D_, D_);
      gemm_wmma<0><<<ggrid(8192, D_), 256, 0, stream>>>(BUF_H, ffW2 + (size_t)l*4*D_*D_, ffb2 + l*D_, nullptr, BUF_D + (size_t)ch*8192*D_, 8192, D_, 4*D_);
    }
    ln_kernel<<<TOK, 256, 0, stream>>>(X, BUF_D, X, n2g + l*D_, n2b + l*D_);
  }
  ln_kernel<<<TOK, 256, 0, stream>>>(X, nullptr, X, fng, fnb);

  // 4. last hidden state
  last_kernel<<<B_, 256, 0, stream>>>(seq, X, LAST);

  // 5. sem & eco_s heads
  gemm_wmma<1><<<ggrid(B_, D_), 256, 0, stream>>>(LAST, semW1, semb1, nullptr, SB1, B_, D_, D_);
  gemm_wmma<0><<<ggrid(B_, SEM_), 256, 0, stream>>>(SB1, semW2, semb2, nullptr, SEMV, B_, SEM_, D_);
  gemm_wmma<1><<<ggrid(B_, D_), 256, 0, stream>>>(LAST, ecoW1, ecob1, nullptr, SB1, B_, D_, D_);
  gemm_wmma<0><<<ggrid(B_, ECO_), 256, 0, stream>>>(SB1, ecoW2, ecob2, nullptr, ECOS, B_, ECO_, D_);

  // 6. candidate embeddings + split concat-matmuls
  embed_kernel<<<B_*C_, 256, 0, stream>>>(cand, item_table, nullptr, side_feat,
                                          side_W, side_b, BUF_A, 0, 0);
  gemm_wmma<0><<<ggrid(B_*C_, D_), 256, 0, stream>>>(BUF_A, doW, nullptr, nullptr, BUF_B, B_*C_, D_, D_);
  gemm_wmma<0><<<ggrid(B_*C_, D_), 256, 0, stream>>>(BUF_A, shW1 + (size_t)SEM_*D_, nullptr, nullptr, BUF_C, B_*C_, D_, D_);
  gemm_wmma<0><<<ggrid(B_, D_), 256, 0, stream>>>(ECOS, doW + (size_t)D_*D_, nullptr, nullptr, ECOWE, B_, D_, ECO_);
  gemm_wmma<0><<<ggrid(B_, D_), 256, 0, stream>>>(SEMV, shW1, nullptr, nullptr, SEMWS, B_, D_, SEM_);
  gemm_wmma<0><<<ggrid(P_, D_), 256, 0, stream>>>(proto, doW + (size_t)(D_+ECO_)*D_, dob, nullptr, PROTOWP, P_, D_, D_);

  // 7. heads + finishers
  do_util_kernel<<<B_*C_, 256, 0, stream>>>(BUF_B, ECOWE, PROTOWP, doOutW, doOutb, plog, OUT_DO);
  short_kernel<<<B_*C_, 256, 0, stream>>>(SEMWS, BUF_C, shb1, shW2, shb2, OUT_SHORT);
  perb_kernel<<<B_, 256, 0, stream>>>(OUT_DO, ECOS, SEMV, refW, refb, etaW, etab,
                                      prop, cand, OUT_REF, OUT_W, DSQ);
  total_kernel<<<(B_*C_ + 255)/256, 256, 0, stream>>>(OUT_DO, OUT_SHORT, OUT_REF, kappa,
                                                      OUT_TOTAL, OUT_LONG);
  ortho_kernel<<<1, 64, 0, stream>>>(DSQ, OUT_ORTHO);
}
